// CausalSelfAttention_3161095930138
// MI455X (gfx1250) — matrix operand order repaired and checked by reading, not compiled, not run on hardware
//
#include <hip/hip_runtime.h>

// ---------------------------------------------------------------------------
// CDNA5 (gfx1250) fused GQA sliding-window attention, bf16 WMMA pipeline.
// All matmuls use v_wmma_f32_16x16x32_bf16 (f32 accumulate).
// GEMM kernels use a 32x64 register tile per wave (8 WMMAs / K-step) to raise
// arithmetic intensity to ~24 B/lane/WMMA of cache traffic.
// ---------------------------------------------------------------------------

typedef __bf16 bf16;
typedef __attribute__((ext_vector_type(16))) __bf16 v16bf;
typedef __attribute__((ext_vector_type(8)))  __bf16 v8bf;
typedef __attribute__((ext_vector_type(8)))  float  v8f;

#define B_    4
#define T_    2048
#define C_    1024
#define H_    16
#define KV_   4
#define HD_   64
#define WIN_  256
#define NREP_ (H_/KV_)
#define QKVO_ ((2*KV_+H_)*HD_)   /* 1536 */

#define NEG_INF (-__builtin_inff())

static __device__ __forceinline__ v8f zero8() {
  v8f z;
#pragma unroll
  for (int i = 0; i < 8; ++i) z[i] = 0.0f;
  return z;
}

// A-fragment (16x32 bf16, M=lane&15): per-lane two contiguous 16B runs.
// group 0 lanes hold K {0..7, 16..23}, group 1 lanes hold K {8..15, 24..31}.
static __device__ __forceinline__ v16bf load_a_frag(const bf16* __restrict__ row, int g) {
  union { v16bf v; v8bf h[2]; } u;
  u.h[0] = *reinterpret_cast<const v8bf*>(row + 8 * g);
  u.h[1] = *reinterpret_cast<const v8bf*>(row + 16 + 8 * g);
  return u.v;
}

static __device__ __forceinline__ v8f wmma_bf16(v16bf a, v16bf b, v8f c) {
  return __builtin_amdgcn_wmma_f32_16x16x32_bf16(false, a, false, b, (short)0, c, false, false);
}

// ---------------------------------------------------------------------------
// Kernel 0a: elementwise f32 -> bf16
// ---------------------------------------------------------------------------
__global__ void cvt_bf16_kernel(const float* __restrict__ in, bf16* __restrict__ out, int n) {
  int i = blockIdx.x * blockDim.x + threadIdx.x;
  int stride = gridDim.x * blockDim.x;
  for (; i < n; i += stride) out[i] = (bf16)in[i];
}

// Kernel 0b: transpose [rows, cols] f32 -> [cols, rows] bf16 (weight prep)
__global__ void cvt_t_bf16_kernel(const float* __restrict__ in, bf16* __restrict__ out,
                                  int rows, int cols) {
  int i = blockIdx.x * blockDim.x + threadIdx.x;
  int n = rows * cols;
  int stride = gridDim.x * blockDim.x;
  for (; i < n; i += stride) {
    int r = i / cols, c = i - r * cols;
    out[(size_t)c * rows + r] = (bf16)in[i];
  }
}

// ---------------------------------------------------------------------------
// Kernel 1: QKV = x * w_attn^T + b, fused RoPE.
//   Q -> qT  [B,H,HD,T]  (transposed: contiguous B-fragments for attention)
//   K -> kb  [B,KV,T,HD] (row-major: A-fragments for S^T = K*Q^T)
//   V -> vb  [B,KV,T,HD] (row-major: B-fragments for O = P*V)
// grid (QKVO/512, T/32, B), block 256 (8 waves, 32x64 tile per wave).
// Wave tile width (64) == head width and all q/k/v segment boundaries are
// 64-aligned, so each wave's epilogue is branch-uniform and single-head.
// ---------------------------------------------------------------------------
__global__ __launch_bounds__(256)
void qkv_rope_kernel(const bf16* __restrict__ xb,     // [B,T,C]
                     const bf16* __restrict__ wT,     // [C, QKVO]
                     const float* __restrict__ bias,  // [QKVO]
                     const float* __restrict__ fcos,  // [T, HD/2]
                     const float* __restrict__ fsin,  // [T, HD/2]
                     bf16* __restrict__ qT,
                     bf16* __restrict__ kb,
                     bf16* __restrict__ vb) {
  const int wave = threadIdx.x >> 5;
  const int lane = threadIdx.x & 31;
  const int g    = lane >> 4;
  const int ln   = lane & 15;

  const int o0w = blockIdx.x * 512 + wave * 64;   // wave's 64 output columns
  const int t0  = blockIdx.y * 32;                // wave's 32 rows
  const int b   = blockIdx.z;

  const bf16* __restrict__ arow0 = xb + (size_t)(b * T_ + t0 + ln) * C_;
  const bf16* __restrict__ arow1 = arow0 + (size_t)16 * C_;
  const int kB = 8 * g;                    // B-fragment K base within 32-chunk

  v8f acc[2][4];
#pragma unroll
  for (int mt = 0; mt < 2; ++mt)
#pragma unroll
    for (int nt = 0; nt < 4; ++nt) acc[mt][nt] = zero8();

#pragma unroll 2
  for (int kk = 0; kk < C_; kk += 32) {
    __builtin_prefetch(arow0 + kk + 128, 0, 0);
    v16bf a0 = load_a_frag(arow0 + kk, g);
    v16bf a1 = load_a_frag(arow1 + kk, g);
    const bf16* __restrict__ brow = wT + (size_t)(kk + kB) * QKVO_ + o0w + ln;
#pragma unroll
    for (int nt = 0; nt < 4; ++nt) {
      v16bf bf;
#pragma unroll
      for (int s = 0; s < 16; ++s) bf[s] = brow[(size_t)(s + (s & 8)) * QKVO_ + nt * 16];
      acc[0][nt] = wmma_bf16(a0, bf, acc[0][nt]);
      acc[1][nt] = wmma_bf16(a1, bf, acc[1][nt]);
    }
  }

  if (o0w < H_ * HD_) {                      // ---- Q segment (RoPE, store Q^T)
    const int h = o0w >> 6;
#pragma unroll
    for (int mt = 0; mt < 2; ++mt) {
#pragma unroll
      for (int nt = 0; nt < 4; ++nt) {
        const int  hd   = nt * 16 + ln;
        const int  j    = hd >> 1;
        const bool even = (hd & 1) == 0;
        const float bv  = bias[o0w + nt * 16 + ln];
        v8bf outv;
#pragma unroll
        for (int r = 0; r < 8; ++r) {
          const int t = t0 + mt * 16 + r + 8 * g;
          float v  = acc[mt][nt][r] + bv;
          float p  = __shfl_xor(v, 1, 32);   // RoPE partner (hd ^ 1)
          float cf = fcos[t * (HD_ / 2) + j];
          float sf = fsin[t * (HD_ / 2) + j];
          float o  = even ? (v * cf - p * sf) : (p * sf + v * cf);
          outv[r] = (bf16)o;
        }
        *reinterpret_cast<v8bf*>(qT + ((size_t)(b * H_ + h) * HD_ + hd) * T_
                                    + t0 + mt * 16 + 8 * g) = outv;
      }
    }
  } else if (o0w < (H_ + KV_) * HD_) {       // ---- K segment (RoPE, row-major)
    const int kv = (o0w - H_ * HD_) >> 6;
#pragma unroll
    for (int mt = 0; mt < 2; ++mt) {
#pragma unroll
      for (int nt = 0; nt < 4; ++nt) {
        const int  hd   = nt * 16 + ln;
        const int  j    = hd >> 1;
        const bool even = (hd & 1) == 0;
        const float bv  = bias[o0w + nt * 16 + ln];
#pragma unroll
        for (int r = 0; r < 8; ++r) {
          const int t = t0 + mt * 16 + r + 8 * g;
          float v  = acc[mt][nt][r] + bv;
          float p  = __shfl_xor(v, 1, 32);
          float cf = fcos[t * (HD_ / 2) + j];
          float sf = fsin[t * (HD_ / 2) + j];
          float o  = even ? (v * cf - p * sf) : (p * sf + v * cf);
          kb[((size_t)(b * KV_ + kv) * T_ + t) * HD_ + hd] = (bf16)o;
        }
      }
    }
  } else {                                   // ---- V segment (row-major)
    const int kv = (o0w - (H_ + KV_) * HD_) >> 6;
#pragma unroll
    for (int mt = 0; mt < 2; ++mt) {
#pragma unroll
      for (int nt = 0; nt < 4; ++nt) {
        const int  hd  = nt * 16 + ln;
        const float bv = bias[o0w + nt * 16 + ln];
#pragma unroll
        for (int r = 0; r < 8; ++r) {
          const int t = t0 + mt * 16 + r + 8 * g;
          vb[((size_t)(b * KV_ + kv) * T_ + t) * HD_ + hd] = (bf16)(acc[mt][nt][r] + bv);
        }
      }
    }
  }
}

// ---------------------------------------------------------------------------
// Kernel 2: sliding-window flash attention.
// Computes S^T = K * Q^T so the post-softmax probabilities sit exactly in the
// A-fragment layout for O = P * V (no cross-lane transpose needed).
// grid (T/64, H, B), block 128 (4 waves; each wave owns 16 query rows).
// ---------------------------------------------------------------------------
__global__ __launch_bounds__(128)
void attn_kernel(const bf16* __restrict__ qT,   // [B,H,HD,T]
                 const bf16* __restrict__ kb,   // [B,KV,T,HD]
                 const bf16* __restrict__ vb,   // [B,KV,T,HD]
                 bf16* __restrict__ yb) {       // [B,T,C]
  const int wave = threadIdx.x >> 5;
  const int lane = threadIdx.x & 31;
  const int g    = lane >> 4;
  const int ln   = lane & 15;

  const int q0  = blockIdx.x * 64 + wave * 16;
  const int h   = blockIdx.y;
  const int b   = blockIdx.z;
  const int kvh = h / NREP_;

  // Q^T B-fragments (lane indexes q = ln, elements gather d in table order)
  const int dB = 8 * g;
  const bf16* __restrict__ qbase = qT + (size_t)(b * H_ + h) * HD_ * T_ + q0;
  v16bf bq0, bq1;
#pragma unroll
  for (int s = 0; s < 16; ++s) {
    bq0[s] = qbase[(size_t)(dB + s + (s & 8)) * T_ + ln];
    bq1[s] = qbase[(size_t)(dB + 32 + s + (s & 8)) * T_ + ln];
  }

  const bf16* __restrict__ krows = kb + (size_t)(b * KV_ + kvh) * T_ * HD_;
  const bf16* __restrict__ vrows = vb + (size_t)(b * KV_ + kvh) * T_ * HD_;

  v8f oacc[4];
#pragma unroll
  for (int i = 0; i < 4; ++i) oacc[i] = zero8();
  float mrun = NEG_INF, lrun = 0.0f;
  const int   q_lane = q0 + ln;
  const float scale  = 0.125f;               // 1/sqrt(64)

  int kb0 = q0 - (WIN_ - 1);
  if (kb0 < 0) kb0 = 0;
  kb0 &= ~31;

  for (int kt = kb0; kt <= q0 + 15; kt += 32) {
    // ---- S^T tiles (keys kt..kt+15 and kt+16..kt+31)
    const bf16* k0row = krows + (size_t)(kt + ln) * HD_;
    const bf16* k1row = k0row + 16 * HD_;
    v8f c0 = zero8(), c1 = zero8();
    c0 = wmma_bf16(load_a_frag(k0row,      g), bq0, c0);
    c0 = wmma_bf16(load_a_frag(k0row + 32, g), bq1, c0);
    c1 = wmma_bf16(load_a_frag(k1row,      g), bq0, c1);
    c1 = wmma_bf16(load_a_frag(k1row + 32, g), bq1, c1);

    // ---- mask + scale; per-lane row stats (lane owns q = ln)
    float s0[8], s1[8];
    float tmax = NEG_INF;
#pragma unroll
    for (int r = 0; r < 8; ++r) {
      const int key0 = kt + r + 8 * g;
      const int key1 = key0 + 16;
      s0[r] = (key0 <= q_lane && key0 > q_lane - WIN_) ? c0[r] * scale : NEG_INF;
      s1[r] = (key1 <= q_lane && key1 > q_lane - WIN_) ? c1[r] * scale : NEG_INF;
      tmax = fmaxf(tmax, fmaxf(s0[r], s1[r]));
    }
    tmax = fmaxf(tmax, __shfl_xor(tmax, 16, 32));
    const float mnew  = fmaxf(mrun, tmax);
    const float muse  = (mnew == NEG_INF) ? 0.0f : mnew;
    const float alpha = (mrun == NEG_INF) ? 0.0f : __expf(mrun - mnew);

    float tsum = 0.0f;
    v16bf pa;                                // P already in A-fragment layout!
#pragma unroll
    for (int r = 0; r < 8; ++r) {
      const float e0 = __expf(s0[r] - muse);
      const float e1 = __expf(s1[r] - muse);
      tsum += e0 + e1;
      pa[r]     = (bf16)e0;
      pa[8 + r] = (bf16)e1;
    }
    tsum += __shfl_xor(tsum, 16, 32);
    lrun = lrun * alpha + tsum;
    mrun = mnew;

    // ---- rescale accumulators (O rows m = r + 8g; alpha lives at lane m)
#pragma unroll
    for (int r = 0; r < 8; ++r) {
      const float ar = __shfl(alpha, r + 8 * g, 32);
#pragma unroll
      for (int dt = 0; dt < 4; ++dt) oacc[dt][r] *= ar;
    }

    // ---- O += P * V   (V B-fragments: lane d = ln, elements gather keys in table order)
    const bf16* vrow = vrows + (size_t)(kt + dB) * HD_ + ln;
#pragma unroll
    for (int dt = 0; dt < 4; ++dt) {
      v16bf bv;
#pragma unroll
      for (int s = 0; s < 16; ++s) bv[s] = vrow[(s + (s & 8)) * HD_ + dt * 16];
      oacc[dt] = wmma_bf16(pa, bv, oacc[dt]);
    }
  }

  // ---- normalize + store y (bf16, [B,T,C] head-major columns)
#pragma unroll
  for (int r = 0; r < 8; ++r) {
    const float lr  = __shfl(lrun, r + 8 * g, 32);
    const float inv = 1.0f / lr;
    const int   t   = q0 + r + 8 * g;
#pragma unroll
    for (int dt = 0; dt < 4; ++dt) {
      yb[(size_t)(b * T_ + t) * C_ + h * HD_ + dt * 16 + ln] = (bf16)(oacc[dt][r] * inv);
    }
  }
}

// ---------------------------------------------------------------------------
// Kernel 3: out = y * w_proj^T + b_proj   (f32 output)
// grid (C/512, T/32, B), block 256 (8 waves, 32x64 tile per wave).
// ---------------------------------------------------------------------------
__global__ __launch_bounds__(256)
void proj_kernel(const bf16* __restrict__ yb,    // [B,T,C]
                 const bf16* __restrict__ wpT,   // [C, C]
                 const float* __restrict__ bias, // [C]
                 float* __restrict__ out) {      // [B,T,C]
  const int wave = threadIdx.x >> 5;
  const int lane = threadIdx.x & 31;
  const int g    = lane >> 4;
  const int ln   = lane & 15;

  const int o0w = blockIdx.x * 512 + wave * 64;
  const int t0  = blockIdx.y * 32;
  const int b   = blockIdx.z;

  const bf16* __restrict__ arow0 = yb + (size_t)(b * T_ + t0 + ln) * C_;
  const bf16* __restrict__ arow1 = arow0 + (size_t)16 * C_;
  const int kB = 8 * g;

  v8f acc[2][4];
#pragma unroll
  for (int mt = 0; mt < 2; ++mt)
#pragma unroll
    for (int nt = 0; nt < 4; ++nt) acc[mt][nt] = zero8();

#pragma unroll 2
  for (int kk = 0; kk < C_; kk += 32) {
    __builtin_prefetch(arow0 + kk + 128, 0, 0);
    v16bf a0 = load_a_frag(arow0 + kk, g);
    v16bf a1 = load_a_frag(arow1 + kk, g);
    const bf16* __restrict__ brow = wpT + (size_t)(kk + kB) * C_ + o0w + ln;
#pragma unroll
    for (int nt = 0; nt < 4; ++nt) {
      v16bf bf;
#pragma unroll
      for (int s = 0; s < 16; ++s) bf[s] = brow[(size_t)(s + (s & 8)) * C_ + nt * 16];
      acc[0][nt] = wmma_bf16(a0, bf, acc[0][nt]);
      acc[1][nt] = wmma_bf16(a1, bf, acc[1][nt]);
    }
  }

#pragma unroll
  for (int mt = 0; mt < 2; ++mt) {
#pragma unroll
    for (int nt = 0; nt < 4; ++nt) {
      const float bv = bias[o0w + nt * 16 + ln];
#pragma unroll
      for (int r = 0; r < 8; ++r) {
        const int t = t0 + mt * 16 + r + 8 * g;
        out[(size_t)(b * T_ + t) * C_ + o0w + nt * 16 + ln] = acc[mt][nt][r] + bv;
      }
    }
  }
}

// ---------------------------------------------------------------------------
// Launch
// inputs: x, freqs_cos, freqs_sin, w_attn, b_attn, w_proj, b_proj (all f32)
// ---------------------------------------------------------------------------
extern "C" void kernel_launch(void* const* d_in, const int* in_sizes, int n_in,
                              void* d_out, int out_size, void* d_ws, size_t ws_size,
                              hipStream_t stream) {
  const float* x      = (const float*)d_in[0];
  const float* fcos   = (const float*)d_in[1];
  const float* fsin   = (const float*)d_in[2];
  const float* w_attn = (const float*)d_in[3];
  const float* b_attn = (const float*)d_in[4];
  const float* w_proj = (const float*)d_in[5];
  const float* b_proj = (const float*)d_in[6];
  float* out = (float*)d_out;

  // workspace carve-up (~61 MB)
  char* ws = (char*)d_ws;
  size_t off = 0;
  bf16* xb  = (bf16*)(ws + off); off += (size_t)B_ * T_ * C_ * 2;
  bf16* wT  = (bf16*)(ws + off); off += (size_t)C_ * QKVO_ * 2;
  bf16* wpT = (bf16*)(ws + off); off += (size_t)C_ * C_ * 2;
  bf16* qT  = (bf16*)(ws + off); off += (size_t)B_ * H_  * T_ * HD_ * 2;
  bf16* kb  = (bf16*)(ws + off); off += (size_t)B_ * KV_ * T_ * HD_ * 2;
  bf16* vb  = (bf16*)(ws + off); off += (size_t)B_ * KV_ * T_ * HD_ * 2;
  bf16* yb  = (bf16*)(ws + off); off += (size_t)B_ * T_ * C_ * 2;
  (void)ws_size; (void)n_in; (void)in_sizes; (void)out_size;

  // 0) precision conversion + weight transposes
  cvt_bf16_kernel  <<<dim3(4096), 256, 0, stream>>>(x, xb, B_ * T_ * C_);
  cvt_t_bf16_kernel<<<dim3(2048), 256, 0, stream>>>(w_attn, wT, QKVO_, C_);
  cvt_t_bf16_kernel<<<dim3(2048), 256, 0, stream>>>(w_proj, wpT, C_, C_);

  // 1) QKV projection + RoPE
  qkv_rope_kernel<<<dim3(QKVO_ / 512, T_ / 32, B_), 256, 0, stream>>>(
      xb, wT, b_attn, fcos, fsin, qT, kb, vb);

  // 2) sliding-window attention
  attn_kernel<<<dim3(T_ / 64, H_, B_), 128, 0, stream>>>(qT, kb, vb, yb);

  // 3) output projection
  proj_kernel<<<dim3(C_ / 512, T_ / 32, B_), 256, 0, stream>>>(yb, wpT, b_proj, out);
}
